// Trans_82154134438209
// MI455X (gfx1250) — matrix operand order repaired and checked by reading, not compiled, not run on hardware
//
#include <hip/hip_runtime.h>
#include <math.h>

typedef __attribute__((ext_vector_type(8)))  _Float16 v8h;
typedef __attribute__((ext_vector_type(16))) _Float16 v16h;
typedef __attribute__((ext_vector_type(8)))  float    v8f;
typedef __attribute__((ext_vector_type(4)))  int      v4i;

#define C_CH  192
#define HW    9216            // 96*96
#define BATCH 4
#define LVLS  5
#define PT    (LVLS*BATCH*HW) // 184320 pixels in Xf
#define PB    (BATCH*HW)      // 36864 pixels at reference scale

#define BM 128
#define BN 96
#define BK 32

#if defined(__has_builtin)
#if __has_builtin(__builtin_amdgcn_global_load_async_to_lds_b128) && \
    __has_builtin(__builtin_amdgcn_s_wait_asynccnt)
#define HAVE_ASYNC_LDS 1
#endif
#endif

#ifdef HAVE_ASYNC_LDS
#define ASYNC_CP16(gsrc, ldst)                                                 \
  __builtin_amdgcn_global_load_async_to_lds_b128((v4i*)(gsrc), (v4i*)(ldst),   \
                                                 0, 0)
#else
#define ASYNC_CP16(gsrc, ldst) (*(v8h*)(ldst) = *(const v8h*)(gsrc))
#endif

enum { EPI_NONE = 0, EPI_RELU = 1, EPI_GELU = 2, EPI_RESBN = 3 };

// ---------------------------------------------------------------------------
// Bilinear resize (half-pixel, clamped) + pack into f16 [pixel, channel]
// ---------------------------------------------------------------------------
__global__ __launch_bounds__(256) void resize_pack_kernel(
    const float* __restrict__ x, int hin, int win,
    _Float16* __restrict__ Xf, int level)
{
  long long idx = (long long)blockIdx.x * 256 + threadIdx.x;
  const long long total = (long long)BATCH * HW * C_CH;
  if (idx >= total) return;
  int c        = (int)(idx % C_CH);
  long long px = idx / C_CH;
  int n  = (int)(px / HW);
  int pp = (int)(px % HW);
  int py = pp / 96, pxx = pp % 96;

  float sy = hin * (1.0f / 96.0f), sx = win * (1.0f / 96.0f);
  float fy = (py  + 0.5f) * sy - 0.5f;
  float fx = (pxx + 0.5f) * sx - 0.5f;
  fy = fminf(fmaxf(fy, 0.0f), (float)(hin - 1));
  fx = fminf(fmaxf(fx, 0.0f), (float)(win - 1));
  int y0 = (int)fy, x0 = (int)fx;
  int y1 = min(y0 + 1, hin - 1), x1 = min(x0 + 1, win - 1);
  float wy = fy - (float)y0, wx = fx - (float)x0;

  const float* base = x + (long long)(n * C_CH + c) * hin * win;
  float v00 = base[y0 * win + x0], v01 = base[y0 * win + x1];
  float v10 = base[y1 * win + x0], v11 = base[y1 * win + x1];
  float v = v00 * (1.0f - wy) * (1.0f - wx) + v01 * (1.0f - wy) * wx
          + v10 * wy * (1.0f - wx)          + v11 * wy * wx;

  Xf[((long long)(level * BATCH + n) * HW + pp) * C_CH + c] = (_Float16)v;
}

// ---------------------------------------------------------------------------
// f32 -> f16 weight conversion
// ---------------------------------------------------------------------------
__global__ __launch_bounds__(256) void cvt_f16_kernel(
    const float* __restrict__ s, _Float16* __restrict__ d, int n)
{
  int i = blockIdx.x * 256 + threadIdx.x;
  if (i < n) d[i] = (_Float16)s[i];
}

// ---------------------------------------------------------------------------
// WMMA GEMM: Out[m, n] = act( sum_k A[m,k] * W[n,k] + bias[n] )
//   Tile: 128x96x32, 8 waves (4M x 2N), 6 wmma / wave / K-step.
//   cat_fuse K remap handled with division-free kBase/lvlOff increments.
//   EPI_RESBN: out = relu((resIn[m,n] + acc + bias) * bnS[n] + bnT[n])
//   out32 (EPI_NONE only): also store f32 NCHW (final output)
// ---------------------------------------------------------------------------
template <int EPI>
__global__ __launch_bounds__(256) void gemm_wmma_kernel(
    const _Float16* __restrict__ A, int lda, int Kc, long long lvlStride,
    const _Float16* __restrict__ Bw, int ldb,
    const float* __restrict__ bias,
    _Float16* __restrict__ Out, int ldo,
    int K,
    const _Float16* __restrict__ resIn,
    const float* __restrict__ bnS, const float* __restrict__ bnT,
    float* __restrict__ out32)
{
  __shared__ _Float16 As[BM * BK];  // [row][k], k contiguous
  __shared__ _Float16 Bs[BN * BK];  // [cout][k], k contiguous

  const int tid  = threadIdx.x;
  const int wave = tid >> 5;
  const int lane = tid & 31;
  const int wm   = wave & 3;   // 4 waves along M (32 rows each)
  const int wn   = wave >> 2;  // 2 waves along N (48 cols each)
  const int half = lane >> 4;  // 0: lanes 0-15, 1: lanes 16-31
  const int l16  = lane & 15;

  const long long m0 = (long long)blockIdx.x * BM;
  const int       n0 = blockIdx.y * BN;

  v8f acc[2][3] = {};

  // division-free K remap state: kBase = k0 % Kc, lvlOff = (k0/Kc)*lvlStride
  int       kBase  = 0;
  long long lvlOff = 0;

  for (int k0 = 0; k0 < K; k0 += BK) {
    const _Float16* Ak = A + kBase + lvlOff;   // uniform per K-step

    // ---- stage A tile: 128x32 halves = 512 x 16B chunks over 256 threads
#pragma unroll
    for (int i = 0; i < 2; ++i) {
      int ch  = tid + i * 256;
      int row = ch >> 2;
      int kc  = (ch & 3) * 8;
      ASYNC_CP16(Ak + (m0 + row) * (long long)lda + kc, &As[row * BK + kc]);
    }
    // ---- stage B tile: 96x32 halves = 384 x 16B chunks
#pragma unroll
    for (int i = 0; i < 2; ++i) {
      int ch = tid + i * 256;
      if (ch < 384) {
        int row = ch >> 2;
        int kc  = (ch & 3) * 8;
        ASYNC_CP16(Bw + (long long)(n0 + row) * ldb + k0 + kc,
                   &Bs[row * BK + kc]);
      }
    }
    // ---- prefetch next A step (division-free)
    if (k0 + BK < K) {
      int       kBaseN  = kBase + BK;
      long long lvlOffN = lvlOff;
      if (kBaseN == Kc) { kBaseN = 0; lvlOffN += lvlStride; }
      const _Float16* An = A + kBaseN + lvlOffN;
      __builtin_prefetch(An + (m0 + (tid >> 1)) * (long long)lda, 0, 1);
    }
#ifdef HAVE_ASYNC_LDS
    __builtin_amdgcn_s_wait_asynccnt(0);
#endif
    __syncthreads();

    // ---- B fragments: lane holds column, K slice half*16..+15
    v16h bfr[3];
#pragma unroll
    for (int s = 0; s < 3; ++s) {
      const _Float16* bp = &Bs[(wn * 48 + s * 16 + l16) * BK + half * 16];
      v8h blo = *(const v8h*)bp;
      v8h bhi = *(const v8h*)(bp + 8);
      bfr[s] = __builtin_shufflevector(blo, bhi, 0, 1, 2, 3, 4, 5, 6, 7,
                                       8, 9, 10, 11, 12, 13, 14, 15);
    }
    // ---- A fragments: lane holds row, K slice half*16..+15
    v16h afr[2];
#pragma unroll
    for (int t = 0; t < 2; ++t) {
      const _Float16* ap = &As[(wm * 32 + t * 16 + l16) * BK];
      v8h alo = *(const v8h*)(ap + half * 16);
      v8h ahi = *(const v8h*)(ap + half * 16 + 8);
      afr[t] = __builtin_shufflevector(alo, ahi, 0, 1, 2, 3, 4, 5, 6, 7,
                                       8, 9, 10, 11, 12, 13, 14, 15);
    }
#pragma unroll
    for (int t = 0; t < 2; ++t)
#pragma unroll
      for (int s = 0; s < 3; ++s)
        acc[t][s] = __builtin_amdgcn_wmma_f32_16x16x32_f16(
            false, afr[t], false, bfr[s], (short)0, acc[t][s], false, false);
    __syncthreads();

    kBase += BK;
    if (kBase == Kc) { kBase = 0; lvlOff += lvlStride; }
  }

  // ---- epilogue: lane (half,l16): col = n0 + wn*48 + s*16 + l16,
  //      VGPR v -> row m0 + wm*32 + t*16 + half*8 + v
#pragma unroll
  for (int t = 0; t < 2; ++t) {
#pragma unroll
    for (int s = 0; s < 3; ++s) {
      const int   col  = n0 + wn * 48 + s * 16 + l16;
      const float bcol = bias[col];
      long long mbase = m0 + wm * 32 + t * 16 + half * 8;
#pragma unroll
      for (int v = 0; v < 8; ++v) {
        long long m = mbase + v;
        float xv = acc[t][s][v] + bcol;
        if (EPI == EPI_RELU) xv = fmaxf(xv, 0.0f);
        if (EPI == EPI_GELU) xv = 0.5f * xv * (1.0f + erff(xv * 0.70710678118f));
        if (EPI == EPI_RESBN) {
          xv = ((float)resIn[m * C_CH + col] + xv) * bnS[col] + bnT[col];
          xv = fmaxf(xv, 0.0f);
        }
        Out[m * (long long)ldo + col] = (_Float16)xv;
        if (EPI == EPI_NONE && out32 != nullptr) {
          int img = (int)(m / HW);
          int p   = (int)(m % HW);
          out32[((long long)img * C_CH + col) * HW + p] = xv;
        }
      }
    }
  }
}

// ---------------------------------------------------------------------------
// Attention: 16 lanes cooperate per (pixel, key-head j). Each lane owns 4
// channels. 15 scores via shfl_xor reduction, softmax, weighted V sum.
// ---------------------------------------------------------------------------
__global__ __launch_bounds__(256) void attn_kernel(
    const _Float16* __restrict__ Key, const _Float16* __restrict__ Q,
    const _Float16* __restrict__ V, _Float16* __restrict__ Ref)
{
  int gid   = blockIdx.x * 256 + threadIdx.x;
  int grp   = gid >> 4;          // one group of 16 lanes per (pixel, j)
  int lane4 = gid & 15;          // channel sub-lane: channels lane4*4 .. +3
  int pixel = grp / 3;
  int j     = grp % 3;
  if (pixel >= PB) return;
  int n = pixel / HW;
  int p = pixel % HW;

  const _Float16* kp = Key + (long long)pixel * C_CH + j * 64 + lane4 * 4;
  float k0f = (float)kp[0], k1f = (float)kp[1], k2f = (float)kp[2], k3f = (float)kp[3];

  float s[15];
#pragma unroll
  for (int l = 0; l < LVLS; ++l) {
    long long base = ((long long)(l * BATCH + n) * HW + p) * C_CH;
#pragma unroll
    for (int t = 0; t < 3; ++t) {
      const _Float16* qp = Q + base + t * 64 + lane4 * 4;
      float d = k0f * (float)qp[0] + k1f * (float)qp[1] +
                k2f * (float)qp[2] + k3f * (float)qp[3];
      d += __shfl_xor(d, 1, 32);
      d += __shfl_xor(d, 2, 32);
      d += __shfl_xor(d, 4, 32);
      d += __shfl_xor(d, 8, 32);
      s[l * 3 + t] = d;
    }
  }

  float mx = s[0];
#pragma unroll
  for (int i = 1; i < 15; ++i) mx = fmaxf(mx, s[i]);
  float sum = 0.0f;
#pragma unroll
  for (int i = 0; i < 15; ++i) { s[i] = __expf(s[i] - mx); sum += s[i]; }
  float inv = 1.0f / sum;

  float a0 = 0.0f, a1 = 0.0f, a2 = 0.0f, a3 = 0.0f;
#pragma unroll
  for (int l = 0; l < LVLS; ++l) {
    long long base = ((long long)(l * BATCH + n) * HW + p) * C_CH;
#pragma unroll
    for (int t = 0; t < 3; ++t) {
      float w = s[l * 3 + t] * inv;
      const _Float16* vp = V + base + t * 64 + lane4 * 4;
      a0 += w * (float)vp[0]; a1 += w * (float)vp[1];
      a2 += w * (float)vp[2]; a3 += w * (float)vp[3];
    }
  }
  _Float16* rp = Ref + (long long)pixel * C_CH + j * 64 + lane4 * 4;
  rp[0] = (_Float16)a0; rp[1] = (_Float16)a1;
  rp[2] = (_Float16)a2; rp[3] = (_Float16)a3;
}

// ---------------------------------------------------------------------------
extern "C" void kernel_launch(void* const* d_in, const int* in_sizes, int n_in,
                              void* d_out, int out_size, void* d_ws, size_t ws_size,
                              hipStream_t stream)
{
  (void)in_sizes; (void)n_in; (void)out_size; (void)ws_size;

  const float* x[5];
  for (int i = 0; i < 5; ++i) x[i] = (const float*)d_in[i];
  const float* W_k1 = (const float*)d_in[5];  const float* b_k1 = (const float*)d_in[6];
  const float* W_k2 = (const float*)d_in[7];  const float* b_k2 = (const float*)d_in[8];
  const float* W_k3 = (const float*)d_in[9];  const float* b_k3 = (const float*)d_in[10];
  const float* W_q  = (const float*)d_in[11]; const float* b_q  = (const float*)d_in[12];
  const float* W_v  = (const float*)d_in[13]; const float* b_v  = (const float*)d_in[14];
  const float* W_cf = (const float*)d_in[15]; const float* b_cf = (const float*)d_in[16];
  const float* W_e1 = (const float*)d_in[17]; const float* b_e1 = (const float*)d_in[18];
  const float* W_e2 = (const float*)d_in[19]; const float* b_e2 = (const float*)d_in[20];
  const float* W_r1 = (const float*)d_in[21]; const float* b_r1 = (const float*)d_in[22];
  const float* W_r2 = (const float*)d_in[23]; const float* b_r2 = (const float*)d_in[24];
  const float* bn_s = (const float*)d_in[25]; const float* bn_t = (const float*)d_in[26];

  char*  ws  = (char*)d_ws;
  size_t off = 0;
  auto alloc = [&](size_t bytes) -> char* {
    char* pp = ws + off;
    off += (bytes + 255) & ~(size_t)255;
    return pp;
  };

  _Float16* Xf  = (_Float16*)alloc((size_t)PT * C_CH * 2);
  _Float16* T1  = (_Float16*)alloc((size_t)PT * 96 * 2);
  _Float16* Qb  = (_Float16*)alloc((size_t)PT * C_CH * 2);
  _Float16* Vb  = (_Float16*)alloc((size_t)PT * C_CH * 2);
  _Float16* CF  = (_Float16*)alloc((size_t)PB * C_CH * 2);
  _Float16* Eb  = (_Float16*)alloc((size_t)PB * 96 * 2);
  _Float16* La  = (_Float16*)alloc((size_t)PB * C_CH * 2);
  _Float16* Kb  = (_Float16*)alloc((size_t)PB * C_CH * 2);
  _Float16* Rf  = (_Float16*)alloc((size_t)PB * C_CH * 2);
  _Float16* wq  = (_Float16*)alloc(36864 * 2);
  _Float16* wv  = (_Float16*)alloc(36864 * 2);
  _Float16* wcf = (_Float16*)alloc(184320 * 2);
  _Float16* wkc = (_Float16*)alloc(36864 * 2);
  _Float16* we1 = (_Float16*)alloc(18432 * 2);
  _Float16* we2 = (_Float16*)alloc(18432 * 2);
  _Float16* wr1 = (_Float16*)alloc(18432 * 2);
  _Float16* wr2 = (_Float16*)alloc(18432 * 2);
  float*    bkc = (float*)alloc(192 * 4);

  auto cvt = [&](const float* s, _Float16* d, int n) {
    cvt_f16_kernel<<<(n + 255) / 256, 256, 0, stream>>>(s, d, n);
  };
  cvt(W_q, wq, 36864);
  cvt(W_v, wv, 36864);
  cvt(W_cf, wcf, 184320);
  cvt(W_k1, wkc, 12288);
  cvt(W_k2, wkc + 12288, 12288);
  cvt(W_k3, wkc + 24576, 12288);
  cvt(W_e1, we1, 18432);
  cvt(W_e2, we2, 18432);
  cvt(W_r1, wr1, 18432);
  cvt(W_r2, wr2, 18432);
  (void)hipMemcpyAsync(bkc,       b_k1, 64 * 4, hipMemcpyDeviceToDevice, stream);
  (void)hipMemcpyAsync(bkc + 64,  b_k2, 64 * 4, hipMemcpyDeviceToDevice, stream);
  (void)hipMemcpyAsync(bkc + 128, b_k3, 64 * 4, hipMemcpyDeviceToDevice, stream);

  const int hs[5] = {192, 96, 48, 24, 12};
  const int rp_blocks = (int)(((long long)PB * C_CH + 255) / 256);
  for (int l = 0; l < 5; ++l)
    resize_pack_kernel<<<rp_blocks, 256, 0, stream>>>(x[l], hs[l], hs[l], Xf, l);

  auto gemm = [&](int epi, const _Float16* A, int lda, int Kc, long long lvl,
                  const _Float16* Bw, int ldb, const float* bias,
                  _Float16* Out, int ldo, long long M, int N, int K,
                  const _Float16* resIn, const float* bnS, const float* bnT,
                  float* o32) {
    dim3 grid((unsigned)(M / BM), (unsigned)(N / BN));
    switch (epi) {
      case EPI_NONE:
        gemm_wmma_kernel<EPI_NONE><<<grid, 256, 0, stream>>>(
            A, lda, Kc, lvl, Bw, ldb, bias, Out, ldo, K, resIn, bnS, bnT, o32);
        break;
      case EPI_RELU:
        gemm_wmma_kernel<EPI_RELU><<<grid, 256, 0, stream>>>(
            A, lda, Kc, lvl, Bw, ldb, bias, Out, ldo, K, resIn, bnS, bnT, o32);
        break;
      case EPI_GELU:
        gemm_wmma_kernel<EPI_GELU><<<grid, 256, 0, stream>>>(
            A, lda, Kc, lvl, Bw, ldb, bias, Out, ldo, K, resIn, bnS, bnT, o32);
        break;
      default:
        gemm_wmma_kernel<EPI_RESBN><<<grid, 256, 0, stream>>>(
            A, lda, Kc, lvl, Bw, ldb, bias, Out, ldo, K, resIn, bnS, bnT, o32);
        break;
    }
  };

  const long long LVL_STRIDE = (long long)PB * C_CH;  // Xf elements per level

  for (int k = 0; k < 5; ++k) {
    // r = relu(Xf * W_r1^T)  ->  Xf = relu((Xf + r*W_r2^T + b)*s + t)
    gemm(EPI_RELU, Xf, C_CH, C_CH, 0, wr1, C_CH, b_r1, T1, 96, PT, 96, C_CH,
         nullptr, nullptr, nullptr, nullptr);
    gemm(EPI_RESBN, T1, 96, 96, 0, wr2, 96, b_r2, Xf, C_CH, PT, C_CH, 96,
         Xf, bn_s, bn_t, nullptr);
    // Q, V projections
    gemm(EPI_NONE, Xf, C_CH, C_CH, 0, wq, C_CH, b_q, Qb, C_CH, PT, C_CH, C_CH,
         nullptr, nullptr, nullptr, nullptr);
    gemm(EPI_NONE, Xf, C_CH, C_CH, 0, wv, C_CH, b_v, Vb, C_CH, PT, C_CH, C_CH,
         nullptr, nullptr, nullptr, nullptr);
    if (k == 0) {
      // cat_fuse over implicit 960-channel concat (K remap) + emb
      gemm(EPI_RELU, Xf, C_CH, C_CH, LVL_STRIDE, wcf, LVLS * C_CH, b_cf,
           CF, C_CH, PB, C_CH, LVLS * C_CH, nullptr, nullptr, nullptr, nullptr);
      gemm(EPI_GELU, CF, C_CH, C_CH, 0, we1, C_CH, b_e1, Eb, 96, PB, 96, C_CH,
           nullptr, nullptr, nullptr, nullptr);
      gemm(EPI_NONE, Eb, 96, 96, 0, we2, 96, b_e2, La, C_CH, PB, C_CH, 96,
           nullptr, nullptr, nullptr, nullptr);
    }
    // 3 key heads fused into one GEMM
    gemm(EPI_NONE, La, C_CH, C_CH, 0, wkc, C_CH, bkc, Kb, C_CH, PB, C_CH, C_CH,
         nullptr, nullptr, nullptr, nullptr);
    // softmax attention over 15 pyramid keys
    attn_kernel<<<(PB * 3 * 16) / 256, 256, 0, stream>>>(Kb, Qb, Vb, Rf);
    // la_res = emb(refs); final iteration also writes f32 NCHW output
    gemm(EPI_GELU, Rf, C_CH, C_CH, 0, we1, C_CH, b_e1, Eb, 96, PB, 96, C_CH,
         nullptr, nullptr, nullptr, nullptr);
    gemm(EPI_NONE, Eb, 96, 96, 0, we2, 96, b_e2, La, C_CH, PB, C_CH, 96,
         nullptr, nullptr, nullptr, (k == 4) ? (float*)d_out : nullptr);
  }
}
